// MPNN_22308060135895
// MI455X (gfx1250) — compile-verified, matrix operand order checked by reading, not run on hardware
//
#include <hip/hip_runtime.h>
#include <hip/hip_fp16.h>

typedef __attribute__((ext_vector_type(8)))  _Float16 v8h;
typedef __attribute__((ext_vector_type(16))) _Float16 v16h;
typedef __attribute__((ext_vector_type(8)))  float    v8f;

#define NNODES 100000
#define NEDGES 1600000
#define HDIM   128
#define NQ     8192

// ---------------- degree / normalization ----------------

__global__ void init_deg_kernel(int* deg, int n) {
    int i = blockIdx.x * blockDim.x + threadIdx.x;
    if (i < n) deg[i] = 1;   // self loop
}

__global__ void count_deg_kernel(const int* __restrict__ dst, int* deg, int e) {
    int i = blockIdx.x * blockDim.x + threadIdx.x;
    if (i < e) atomicAdd(&deg[dst[i]], 1);
}

__global__ void dinv_kernel(float* buf, int n) {
    int i = blockIdx.x * blockDim.x + threadIdx.x;
    if (i < n) {
        int d = ((const int*)buf)[i];
        buf[i] = rsqrtf((float)d);   // d >= 1 always
    }
}

// ---------------- operand prep ----------------

// f32 row-major W[k][n] (128x128)  ->  f16 column-major Wt[n][k]
__global__ void pack_weight_kernel(const float* __restrict__ W, _Float16* __restrict__ Wt) {
    int i = blockIdx.x * blockDim.x + threadIdx.x;
    if (i >= HDIM * HDIM) return;
    const int n = i >> 7, k = i & (HDIM - 1);
    Wt[i] = (_Float16)W[k * HDIM + n];
}

__global__ void f32_to_f16_kernel(const float* __restrict__ s, _Float16* __restrict__ d, size_t n) {
    size_t i = (size_t)blockIdx.x * blockDim.x + threadIdx.x;
    if (i < n) d[i] = (_Float16)s[i];
}

// ---------------- WMMA GEMM: Out[M x 128] = A[M x 128] @ W[128 x 128] ----------------
// A: f16 row-major; Wt: f16 column-major (staged to LDS once per block).
// One wave per 16x16 output tile; K loop = 4x v_wmma_f32_16x16x32_f16.

__global__ __launch_bounds__(256)
void gemm_wmma_kernel(const _Float16* __restrict__ A, const _Float16* __restrict__ Wt,
                      float* __restrict__ Out, int M,
                      const float* __restrict__ bias, int do_relu) {
    __shared__ _Float16 lw[HDIM * HDIM];   // 32 KB of f16 weights, col-major

    // cooperative stage: 2048 x 16B chunks across 256 threads
    {
        const uint4* s = (const uint4*)Wt;
        uint4*       d = (uint4*)lw;
        #pragma unroll
        for (int i = 0; i < (HDIM * HDIM) / (8 * 256); ++i)
            d[i * 256 + threadIdx.x] = s[i * 256 + threadIdx.x];
    }
    __syncthreads();

    const int lane = threadIdx.x & 31;
    const int wave = threadIdx.x >> 5;
    const int tile = blockIdx.x * 8 + wave;
    if (tile >= (M / 16) * 8) return;                  // wave-uniform

    const int tm = tile >> 3, tn = tile & 7;
    const int row0 = tm * 16, col0 = tn * 16;
    const int mr   = lane & 15;                        // A row / B col within tile
    const int khalf = (lane >> 4) * 8;

    const _Float16* __restrict__ arow = A  + (size_t)(row0 + mr) * HDIM + khalf;
    const _Float16*              brow = lw + (size_t)(col0 + mr) * HDIM + khalf;

    v8f acc = {};
    #pragma unroll
    for (int k0 = 0; k0 < HDIM; k0 += 32) {
        union { v16h v; v8h h[2]; } af, bf;
        af.h[0] = *(const v8h*)(arow + k0);
        af.h[1] = *(const v8h*)(arow + k0 + 16);
        bf.h[0] = *(const v8h*)(brow + k0);
        bf.h[1] = *(const v8h*)(brow + k0 + 16);
        acc = __builtin_amdgcn_wmma_f32_16x16x32_f16(
                  false, af.v, false, bf.v, (short)0, acc, false, false);
    }

    const int orow = row0 + (lane >> 4) * 8;
    const int ocol = col0 + (lane & 15);
    const float bv = bias ? bias[ocol] : 0.0f;
    #pragma unroll
    for (int r = 0; r < 8; ++r) {
        float x = acc[r] + bv;
        if (do_relu) x = fmaxf(x, 0.0f);
        Out[(size_t)(orow + r) * HDIM + ocol] = x;
    }
}

// ---------------- scatter / finalize ----------------

__global__ void zero_kernel(float* p, size_t n) {
    size_t i = (size_t)blockIdx.x * blockDim.x + threadIdx.x;
    if (i < n) p[i] = 0.0f;
}

// one wave per edge; lane handles a float4 column chunk; L2-resident f32 atomics
__global__ void scatter_kernel(const float* __restrict__ T,
                               const int* __restrict__ src,
                               const int* __restrict__ dst,
                               const float* __restrict__ dinv,
                               float* __restrict__ acc, int e) {
    long long gid = (long long)blockIdx.x * blockDim.x + threadIdx.x;
    int ed   = (int)(gid >> 5);
    int lane = (int)(gid & 31);
    if (ed >= e) return;
    const int s = src[ed], d = dst[ed];
    const float nrm = dinv[s] * dinv[d];
    const float4 v = *(const float4*)(T + (size_t)s * HDIM + lane * 4);
    float* o = acc + (size_t)d * HDIM + lane * 4;
    unsafeAtomicAdd(o + 0, v.x * nrm);
    unsafeAtomicAdd(o + 1, v.y * nrm);
    unsafeAtomicAdd(o + 2, v.z * nrm);
    unsafeAtomicAdd(o + 3, v.w * nrm);
}

// hout(f16) = act(acc + T * dinv^2 + b)   (self-loop + bias + optional relu)
__global__ void finalize_kernel(const float* __restrict__ acc,
                                const float* __restrict__ T,
                                const float* __restrict__ dinv,
                                const float* __restrict__ bias,
                                int do_relu, _Float16* __restrict__ hout, int n) {
    size_t i = (size_t)blockIdx.x * blockDim.x + threadIdx.x;
    if (i >= (size_t)n * HDIM) return;
    const int row = (int)(i >> 7);
    const int col = (int)(i & (HDIM - 1));
    const float di = dinv[row];
    float x = acc[i] + T[i] * di * di + bias[col];
    if (do_relu) x = fmaxf(x, 0.0f);
    hout[i] = (_Float16)x;
}

// ---------------- link head ----------------

__global__ void link_gather_kernel(const _Float16* __restrict__ h,
                                   const int* __restrict__ root,
                                   _Float16* __restrict__ lk) {
    size_t i = (size_t)blockIdx.x * blockDim.x + threadIdx.x;
    if (i >= (size_t)NQ * HDIM) return;
    const int q   = (int)(i >> 7);
    const int col = (int)(i & (HDIM - 1));
    const int a = root[2 * q];
    const int b = root[2 * q + 1];
    const float p = (float)h[(size_t)a * HDIM + col] * (float)h[(size_t)b * HDIM + col];
    lk[i] = (_Float16)p;
}

// logits[q] = dot(Z[q,:], P2) + pb2 ; one wave per query
__global__ void gemv_kernel(const float* __restrict__ Z,
                            const float* __restrict__ P2,
                            const float* __restrict__ pb2,
                            float* __restrict__ out) {
    long long gid = (long long)blockIdx.x * blockDim.x + threadIdx.x;
    int q    = (int)(gid >> 5);
    int lane = (int)(gid & 31);
    if (q >= NQ) return;
    const float4 z = *(const float4*)(Z + (size_t)q * HDIM + lane * 4);
    const float4 p = *(const float4*)(P2 + lane * 4);
    float s = z.x * p.x + z.y * p.y + z.z * p.z + z.w * p.w;
    #pragma unroll
    for (int off = 16; off > 0; off >>= 1) s += __shfl_xor(s, off, 32);
    if (lane == 0) out[q] = s + pb2[0];
}

// ---------------- launch ----------------

static inline size_t align256(size_t x) { return (x + 255) & ~(size_t)255; }

extern "C" void kernel_launch(void* const* d_in, const int* in_sizes, int n_in,
                              void* d_out, int out_size, void* d_ws, size_t ws_size,
                              hipStream_t stream) {
    (void)in_sizes; (void)n_in; (void)out_size; (void)ws_size;

    const float* x    = (const float*)d_in[0];
    const int*   ei   = (const int*)d_in[1];      // [2,E]
    const int*   root = (const int*)d_in[2];      // [B,2]
    const float* W1   = (const float*)d_in[3];
    const float* b1   = (const float*)d_in[4];
    const float* W2   = (const float*)d_in[5];
    const float* b2   = (const float*)d_in[6];
    const float* W3   = (const float*)d_in[7];
    const float* b3   = (const float*)d_in[8];
    const float* P1   = (const float*)d_in[9];
    const float* pb1  = (const float*)d_in[10];
    const float* P2   = (const float*)d_in[11];
    const float* pb2  = (const float*)d_in[12];
    float* logits = (float*)d_out;

    const int* src = ei;
    const int* dst = ei + NEDGES;

    // workspace carve-up
    char* ws = (char*)d_ws;
    size_t off = 0;
    float*    dinv = (float*)   (ws + off); off += align256((size_t)NNODES * 4);
    float*    T    = (float*)   (ws + off); off += align256((size_t)NNODES * HDIM * 4);
    float*    ACC  = (float*)   (ws + off); off += align256((size_t)NNODES * HDIM * 4);
    _Float16* Ha   = (_Float16*)(ws + off); off += align256((size_t)NNODES * HDIM * 2);
    _Float16* Hb   = (_Float16*)(ws + off); off += align256((size_t)NNODES * HDIM * 2);
    _Float16* Wt1  = (_Float16*)(ws + off); off += align256((size_t)HDIM * HDIM * 2);
    _Float16* Wt2  = (_Float16*)(ws + off); off += align256((size_t)HDIM * HDIM * 2);
    _Float16* Wt3  = (_Float16*)(ws + off); off += align256((size_t)HDIM * HDIM * 2);
    _Float16* Pt1  = (_Float16*)(ws + off); off += align256((size_t)HDIM * HDIM * 2);
    _Float16* LKh  = (_Float16*)(ws + off); off += align256((size_t)NQ * HDIM * 2);
    float*    Z    = (float*)   (ws + off); off += align256((size_t)NQ * HDIM * 4);

    const int B256 = 256;
    const size_t NH = (size_t)NNODES * HDIM;
    const int gridNH    = (int)((NH + B256 - 1) / B256);
    const int gridEdge  = (int)(((long long)NEDGES * 32 + B256 - 1) / B256);
    const int gridW     = (HDIM * HDIM + B256 - 1) / B256;
    const int gridGemmN = (NNODES / 16) * 8 / 8;   // 8 waves per block
    const int gridGemmQ = (NQ / 16) * 8 / 8;

    // normalization
    init_deg_kernel <<<(NNODES + B256 - 1) / B256, B256, 0, stream>>>((int*)dinv, NNODES);
    count_deg_kernel<<<(NEDGES + B256 - 1) / B256, B256, 0, stream>>>(dst, (int*)dinv, NEDGES);
    dinv_kernel     <<<(NNODES + B256 - 1) / B256, B256, 0, stream>>>(dinv, NNODES);

    // operand prep (f16 packing)
    pack_weight_kernel<<<gridW, B256, 0, stream>>>(W1, Wt1);
    pack_weight_kernel<<<gridW, B256, 0, stream>>>(W2, Wt2);
    pack_weight_kernel<<<gridW, B256, 0, stream>>>(W3, Wt3);
    pack_weight_kernel<<<gridW, B256, 0, stream>>>(P1, Pt1);
    f32_to_f16_kernel<<<gridNH, B256, 0, stream>>>(x, Ha, NH);

    // layer 1: Ha -> Hb
    gemm_wmma_kernel<<<gridGemmN, B256, 0, stream>>>(Ha, Wt1, T, NNODES, nullptr, 0);
    zero_kernel     <<<gridNH, B256, 0, stream>>>(ACC, NH);
    scatter_kernel  <<<gridEdge, B256, 0, stream>>>(T, src, dst, dinv, ACC, NEDGES);
    finalize_kernel <<<gridNH, B256, 0, stream>>>(ACC, T, dinv, b1, 1, Hb, NNODES);

    // layer 2: Hb -> Ha
    gemm_wmma_kernel<<<gridGemmN, B256, 0, stream>>>(Hb, Wt2, T, NNODES, nullptr, 0);
    zero_kernel     <<<gridNH, B256, 0, stream>>>(ACC, NH);
    scatter_kernel  <<<gridEdge, B256, 0, stream>>>(T, src, dst, dinv, ACC, NEDGES);
    finalize_kernel <<<gridNH, B256, 0, stream>>>(ACC, T, dinv, b2, 1, Ha, NNODES);

    // layer 3: Ha -> Hb (no relu)
    gemm_wmma_kernel<<<gridGemmN, B256, 0, stream>>>(Ha, Wt3, T, NNODES, nullptr, 0);
    zero_kernel     <<<gridNH, B256, 0, stream>>>(ACC, NH);
    scatter_kernel  <<<gridEdge, B256, 0, stream>>>(T, src, dst, dinv, ACC, NEDGES);
    finalize_kernel <<<gridNH, B256, 0, stream>>>(ACC, T, dinv, b3, 0, Hb, NNODES);

    // link head
    link_gather_kernel<<<(int)(((size_t)NQ * HDIM + B256 - 1) / B256), B256, 0, stream>>>(Hb, root, LKh);
    gemm_wmma_kernel  <<<gridGemmQ, B256, 0, stream>>>(LKh, Pt1, Z, NQ, pb1, 1);
    gemv_kernel       <<<(int)(((long long)NQ * 32 + B256 - 1) / B256), B256, 0, stream>>>(Z, P2, pb2, logits);
}
